// BilinearAttention_31576599560651
// MI455X (gfx1250) — matrix operand order repaired and checked by reading, not compiled, not run on hardware
//
#include <hip/hip_runtime.h>
#include <hip/hip_bf16.h>
#include <math.h>

typedef __attribute__((ext_vector_type(16))) __bf16 v16bf;
typedef __attribute__((ext_vector_type(8)))  __bf16 v8bf;
typedef __attribute__((ext_vector_type(4)))  __bf16 v4bf;
typedef __attribute__((ext_vector_type(8)))  float  v8f;

#define D_Q   1024
#define D_T   1024
#define KDIM  1024
#define G_CNT 4
#define BS    32
#define M_LEN 2048
#define NROWS (BS * M_LEN)     // 65536
#define EPS   1e-5f

// ---------------------------------------------------------------------------
// f32 -> bf16 convert (for V matrices so the GEMM streams bf16 B-fragments)
// ---------------------------------------------------------------------------
__global__ __launch_bounds__(256) void cvt_bf16_kernel(
    const float* __restrict__ in, __bf16* __restrict__ out, int n4) {
  int i = blockIdx.x * 256 + threadIdx.x;
  if (i < n4) {
    float4 f = ((const float4*)in)[i];
    v4bf o;
    o[0] = (__bf16)f.x; o[1] = (__bf16)f.y; o[2] = (__bf16)f.z; o[3] = (__bf16)f.w;
    ((v4bf*)out)[i] = o;
  }
}

// ---------------------------------------------------------------------------
// Big GEMM:  out[r,c] = relu( sum_t Y[r,t] * V[c,t] ),  r<65536, c<1024, t<1024
// One workgroup = 128 rows. Whole 128x1024 Y strip staged in LDS as bf16
// (258 KB, uses CDNA5 320KB LDS) so Y is read from HBM exactly once.
// 8 waves tiled 4(row-groups) x 2(col-groups); each wave: 2 row-tiles x 4
// col-tiles of 16x16 accumulators -> v_wmma_f32_16x16x32_bf16.
// ---------------------------------------------------------------------------
#define GEMM_BM 128
#define LDS_STR 1032   // bf16 elems per LDS row (8-elem pad: 2064B, breaks bank conflicts)

__global__ __launch_bounds__(256) void yv_gemm_relu_kernel(
    const float* __restrict__ Y, const __bf16* __restrict__ Vb,
    __bf16* __restrict__ out) {
  extern __shared__ __bf16 lds[];   // [128][LDS_STR]
  const int tid = threadIdx.x;
  const long rowBlock = (long)blockIdx.x * GEMM_BM;

  // ---- stage 128x1024 f32 -> bf16 into LDS (512 KB read, once) ----
  for (int it = 0; it < 128; ++it) {
    int idx = it * 256 + tid;          // 32768 float4 groups
    int r   = idx >> 8;                // 256 float4 per row
    int c4  = idx & 255;
    float4 f = ((const float4*)(Y + (rowBlock + r) * (long)D_T))[c4];
    __bf16* dst = &lds[r * LDS_STR + c4 * 4];
    dst[0] = (__bf16)f.x; dst[1] = (__bf16)f.y;
    dst[2] = (__bf16)f.z; dst[3] = (__bf16)f.w;
  }
  __syncthreads();

  const int lane  = tid & 31;
  const int w     = tid >> 5;
  const int wr    = w & 3;            // 4 row-groups (2 row-tiles each)
  const int wc    = w >> 2;           // 2 col-groups (64 cols each)
  const int l16   = lane & 15;
  const int hihlf = lane >> 4;        // 0|1
  const int abase = hihlf ? 8 : 0;    // A frag K-base  (ISA 7.12.2, 16-bit A)
  const int bbase = hihlf ? 8 : 0;    // B frag K-base  (ISA 7.12.2, 16-bit B)

  for (int nb = 0; nb < 8; ++nb) {
    const int cb = nb * 128 + wc * 64;
    v8f acc[2][4];
    v8f zero = {0.f, 0.f, 0.f, 0.f, 0.f, 0.f, 0.f, 0.f};
#pragma unroll
    for (int rt = 0; rt < 2; ++rt)
#pragma unroll
      for (int i = 0; i < 4; ++i) acc[rt][i] = zero;

    for (int kt = 0; kt < D_T; kt += 32) {
      v16bf a[2];
#pragma unroll
      for (int rt = 0; rt < 2; ++rt) {
        int row = (wr * 2 + rt) * 16 + l16;
        const __bf16* ap = &lds[row * LDS_STR + kt + abase];
        v8bf alo = *(const v8bf*)ap;
        v8bf ahi = *(const v8bf*)(ap + 16);
        v16bf av;
#pragma unroll
        for (int j = 0; j < 8; ++j) { av[j] = alo[j]; av[j + 8] = ahi[j]; }
        a[rt] = av;
      }
#pragma unroll
      for (int i = 0; i < 4; ++i) {
        int c = cb + i * 16 + l16;
        const __bf16* bp = Vb + (long)c * D_T + kt + bbase;
        v8bf blo = *(const v8bf*)bp;
        v8bf bhi = *(const v8bf*)(bp + 16);
        v16bf bv;
#pragma unroll
        for (int j = 0; j < 8; ++j) { bv[j] = blo[j]; bv[j + 8] = bhi[j]; }
#pragma unroll
        for (int rt = 0; rt < 2; ++rt)
          acc[rt][i] = __builtin_amdgcn_wmma_f32_16x16x32_bf16(
              false, a[rt], false, bv, (short)0, acc[rt][i], false, false);
      }
    }

    // epilogue: relu + bf16 store. D layout: VGPR v -> M = v + 8*(lane>=16)
#pragma unroll
    for (int rt = 0; rt < 2; ++rt)
#pragma unroll
      for (int i = 0; i < 4; ++i)
#pragma unroll
        for (int v = 0; v < 8; ++v) {
          float x = acc[rt][i][v];
          x = x > 0.f ? x : 0.f;
          long row = rowBlock + (wr * 2 + rt) * 16 + hihlf * 8 + v;
          int  col = cb + i * 16 + l16;
          out[row * KDIM + col] = (__bf16)x;
        }
  }
}

// ---------------------------------------------------------------------------
// XU kernel: w_attn[b,k]=relu(f.U_attn[k])*p[g][k]; xu_joint[b,k]=relu(f.U_joint[k])
// grid (32, 8), 256 thr; f row staged in LDS.
// ---------------------------------------------------------------------------
__global__ __launch_bounds__(256) void xu_kernel(
    const float* __restrict__ f, const float* __restrict__ Ua,
    const float* __restrict__ Uj, const float* __restrict__ pg,
    float* __restrict__ w_attn, float* __restrict__ xu_joint) {
  __shared__ float xs[D_Q];
  int b = blockIdx.x, ch = blockIdx.y, t = threadIdx.x;
  for (int i = t; i < D_Q; i += 256) xs[i] = f[b * D_Q + i];
  __syncthreads();
  int k = ch * 128 + (t & 127);
  const float* Urow = ((t < 128) ? Ua : Uj) + (long)k * D_Q;
  float acc = 0.f;
  for (int j = 0; j < D_Q; j += 4) {
    float4 u = *(const float4*)(Urow + j);
    acc += xs[j] * u.x + xs[j + 1] * u.y + xs[j + 2] * u.z + xs[j + 3] * u.w;
  }
  acc = fmaxf(acc, 0.f);
  if (t < 128) w_attn[b * KDIM + k] = acc * pg[k];
  else         xu_joint[b * KDIM + k] = acc;
}

// ---------------------------------------------------------------------------
// logits[b,m] = sum_k w_attn[b,k] * YVa[b,m,k]   (one wave per (b,m))
// ---------------------------------------------------------------------------
__global__ __launch_bounds__(256) void logits_kernel(
    const __bf16* __restrict__ YVa, const float* __restrict__ w_attn,
    float* __restrict__ logits) {
  int wv = threadIdx.x >> 5, lane = threadIdx.x & 31;
  int id = blockIdx.x * 8 + wv;            // 0..65535
  int b = id >> 11, m = id & 2047;
  const __bf16* row = YVa + (long)id * KDIM;
  const float*  wp  = w_attn + b * KDIM;
  float acc = 0.f;
  for (int i = 0; i < 8; ++i) {
    int k = i * 128 + lane * 4;
    v4bf y = *(const v4bf*)(row + k);
    float4 w4 = *(const float4*)(wp + k);
    acc += (float)y[0] * w4.x + (float)y[1] * w4.y +
           (float)y[2] * w4.z + (float)y[3] * w4.w;
  }
  for (int off = 16; off; off >>= 1) acc += __shfl_xor(acc, off, 32);
  if (lane == 0) logits[b * M_LEN + m] = acc;
}

// ---------------------------------------------------------------------------
// softmax over M per batch row; writes normalized attn directly into d_out.
// ---------------------------------------------------------------------------
__global__ __launch_bounds__(256) void softmax_kernel(
    const float* __restrict__ logits, float* __restrict__ attn) {
  __shared__ float red[8];
  int b = blockIdx.x, t = threadIdx.x;
  const float* lrow = logits + b * M_LEN;
  float vals[8];
  float mx = -INFINITY;
  for (int i = 0; i < 8; ++i) { vals[i] = lrow[i * 256 + t]; mx = fmaxf(mx, vals[i]); }
  for (int off = 16; off; off >>= 1) mx = fmaxf(mx, __shfl_xor(mx, off, 32));
  if ((t & 31) == 0) red[t >> 5] = mx;
  __syncthreads();
  float bm = red[0];
  for (int i = 1; i < 8; ++i) bm = fmaxf(bm, red[i]);
  __syncthreads();
  float s = 0.f;
  for (int i = 0; i < 8; ++i) { vals[i] = __expf(vals[i] - bm); s += vals[i]; }
  for (int off = 16; off; off >>= 1) s += __shfl_xor(s, off, 32);
  if ((t & 31) == 0) red[t >> 5] = s;
  __syncthreads();
  float tot = 0.f;
  for (int i = 0; i < 8; ++i) tot += red[i];
  float inv = 1.f / tot;
  for (int i = 0; i < 8; ++i) attn[b * M_LEN + i * 256 + t] = vals[i] * inv;
}

// ---------------------------------------------------------------------------
// Vy_hat partials: partial[b][mc][k] = sum over 256 m of attn[b,m]*YVj[b,m,k]
// grid (32, 8). Fixed-order -> deterministic (no atomics).
// ---------------------------------------------------------------------------
__global__ __launch_bounds__(256) void vyhat_partial_kernel(
    const __bf16* __restrict__ YVj, const float* __restrict__ attn,
    float* __restrict__ partial) {
  int b = blockIdx.x, mc = blockIdx.y, t = threadIdx.x;
  int k = t * 4;
  float a0 = 0.f, a1 = 0.f, a2 = 0.f, a3 = 0.f;
  for (int mm = 0; mm < 256; ++mm) {
    int m = mc * 256 + mm;
    float am = attn[b * M_LEN + m];
    v4bf y = *(const v4bf*)(YVj + ((long)(b * M_LEN + m)) * KDIM + k);
    a0 += am * (float)y[0]; a1 += am * (float)y[1];
    a2 += am * (float)y[2]; a3 += am * (float)y[3];
  }
  float* o = partial + ((b * 8 + mc) * KDIM) + k;
  o[0] = a0; o[1] = a1; o[2] = a2; o[3] = a3;
}

// ---------------------------------------------------------------------------
// Fused: Vy_hat reduce -> h = xu_joint*Vy_hat -> f_joint = h @ P^T
//        -> residual add -> LayerNorm. One block per batch row.
// ---------------------------------------------------------------------------
__global__ __launch_bounds__(256) void fjoint_ln_kernel(
    const float* __restrict__ partial, const float* __restrict__ xu_joint,
    const float* __restrict__ P, const float* __restrict__ f_in,
    const float* __restrict__ gamma, const float* __restrict__ beta,
    float* __restrict__ f_out, float* __restrict__ final_out, int is_last) {
  __shared__ float h[KDIM];
  __shared__ float vbuf[D_Q];
  __shared__ float red[16];
  int b = blockIdx.x, t = threadIdx.x;

  for (int j = 0; j < 4; ++j) {
    int k = t + j * 256;
    float vy = 0.f;
    for (int mc = 0; mc < 8; ++mc) vy += partial[(b * 8 + mc) * KDIM + k];
    h[k] = xu_joint[b * KDIM + k] * vy;
  }
  __syncthreads();

  float sum = 0.f, ssq = 0.f;
  for (int j = 0; j < 4; ++j) {
    int d = t + j * 256;
    const float* Pr = P + (long)d * KDIM;
    float fj = 0.f;
    for (int k = 0; k < KDIM; k += 4) {
      float4 pv = *(const float4*)(Pr + k);
      fj += h[k] * pv.x + h[k + 1] * pv.y + h[k + 2] * pv.z + h[k + 3] * pv.w;
    }
    float v = f_in[b * D_Q + d] + fj;
    vbuf[d] = v;
    sum += v; ssq += v * v;
  }
  for (int off = 16; off; off >>= 1) {
    sum += __shfl_xor(sum, off, 32);
    ssq += __shfl_xor(ssq, off, 32);
  }
  if ((t & 31) == 0) { red[t >> 5] = sum; red[8 + (t >> 5)] = ssq; }
  __syncthreads();
  float S = 0.f, Q = 0.f;
  for (int i = 0; i < 8; ++i) { S += red[i]; Q += red[8 + i]; }
  float mu  = S * (1.f / D_Q);
  float var = Q * (1.f / D_Q) - mu * mu;
  float inv = rsqrtf(var + EPS);
  for (int j = 0; j < 4; ++j) {
    int d = t + j * 256;
    float v = (vbuf[d] - mu) * inv * gamma[d] + beta[d];
    f_out[b * D_Q + d] = v;
    if (is_last) final_out[b * D_Q + d] = v;
  }
}

// ---------------------------------------------------------------------------
extern "C" void kernel_launch(void* const* d_in, const int* in_sizes, int n_in,
                              void* d_out, int out_size, void* d_ws, size_t ws_size,
                              hipStream_t stream) {
  (void)in_sizes; (void)n_in; (void)out_size; (void)ws_size;
  const float* X       = (const float*)d_in[0];
  const float* Y       = (const float*)d_in[1];
  const float* U_attn  = (const float*)d_in[2];
  const float* V_attn  = (const float*)d_in[3];
  const float* p       = (const float*)d_in[4];
  const float* U_joint = (const float*)d_in[5];
  const float* V_joint = (const float*)d_in[6];
  const float* P       = (const float*)d_in[7];
  const float* gamma   = (const float*)d_in[8];
  const float* beta    = (const float*)d_in[9];
  float* out = (float*)d_out;

  char* ws = (char*)d_ws;
  size_t off = 0;
  auto alloc = [&](size_t bytes) -> void* {
    void* q = ws + off;
    off = (off + bytes + 255) & ~(size_t)255;
    return q;
  };
  __bf16* YVa     = (__bf16*)alloc((size_t)NROWS * KDIM * 2);   // 128 MB
  __bf16* YVj     = (__bf16*)alloc((size_t)NROWS * KDIM * 2);   // 128 MB
  __bf16* Vba     = (__bf16*)alloc((size_t)KDIM * D_T * 2);     // 2 MB
  __bf16* Vbj     = (__bf16*)alloc((size_t)KDIM * D_T * 2);     // 2 MB
  float*  f_cur   = (float*)alloc((size_t)BS * D_Q * 4);
  float*  w_attn  = (float*)alloc((size_t)BS * KDIM * 4);
  float*  xu_j    = (float*)alloc((size_t)BS * KDIM * 4);
  float*  logits  = (float*)alloc((size_t)BS * M_LEN * 4);
  float*  partial = (float*)alloc((size_t)BS * 8 * KDIM * 4);   // 1 MB

  cvt_bf16_kernel<<<1024, 256, 0, stream>>>(V_attn, Vba, KDIM * D_T / 4);
  cvt_bf16_kernel<<<1024, 256, 0, stream>>>(V_joint, Vbj, KDIM * D_T / 4);

  const size_t ldsBytes = (size_t)GEMM_BM * LDS_STR * 2;  // 264192 B < 320 KB
  yv_gemm_relu_kernel<<<NROWS / GEMM_BM, 256, ldsBytes, stream>>>(Y, Vba, YVa);
  yv_gemm_relu_kernel<<<NROWS / GEMM_BM, 256, ldsBytes, stream>>>(Y, Vbj, YVj);

  hipMemcpyAsync(f_cur, X, (size_t)BS * D_Q * 4, hipMemcpyDeviceToDevice, stream);

  float* attn_base = out + BS * D_Q;  // attn slab: [G][BS][M]
  for (int g = 0; g < G_CNT; ++g) {
    xu_kernel<<<dim3(32, 8), 256, 0, stream>>>(f_cur, U_attn, U_joint,
                                               p + g * KDIM, w_attn, xu_j);
    logits_kernel<<<NROWS / 8, 256, 0, stream>>>(YVa, w_attn, logits);
    float* attn_g = attn_base + (size_t)g * BS * M_LEN;
    softmax_kernel<<<BS, 256, 0, stream>>>(logits, attn_g);
    vyhat_partial_kernel<<<dim3(32, 8), 256, 0, stream>>>(YVj, attn_g, partial);
    fjoint_ln_kernel<<<BS, 256, 0, stream>>>(partial, xu_j, P, f_cur,
                                             gamma + g * D_Q, beta + g * D_Q,
                                             f_cur, out, g == G_CNT - 1 ? 1 : 0);
  }
}
